// EGNNLayer_30176440221913
// MI455X (gfx1250) — matrix operand order repaired and checked by reading, not compiled, not run on hardware
//
#include <hip/hip_runtime.h>
#include <hip/hip_bf16.h>

#define N_NODES 50000
#define N_EDGES 800000
#define OUT_POS_OFF ((size_t)N_NODES * 128)

typedef __bf16 bf16_t;
typedef bf16_t v16bf __attribute__((ext_vector_type(16)));
typedef float  v8f   __attribute__((ext_vector_type(8)));

union FragAB { unsigned int u[8]; v16bf v; };

static __device__ __forceinline__ unsigned short f2bf(float f) {
    union { float f; unsigned int u; } c; c.f = f;
    unsigned int u = c.u;
    u += 0x7fffu + ((u >> 16) & 1u);   // round-to-nearest-even
    return (unsigned short)(u >> 16);
}
static __device__ __forceinline__ float bf2f(unsigned short h) {
    union { unsigned int u; float f; } c; c.u = ((unsigned int)h) << 16;
    return c.f;
}
static __device__ __forceinline__ unsigned int pack2(float a, float b) {
    return (unsigned int)f2bf(a) | ((unsigned int)f2bf(b) << 16);
}
static __device__ __forceinline__ float siluf(float v) {
    return v * (1.0f / (1.0f + __expf(-v)));
}

// ---------------------------------------------------------------------------
// Core WMMA tile GEMM: 128 rows (LDS, bf16 pair-packed) x [CHUNKS*32, 128]
// weights (global, bf16 pair-packed, row-pair-major). Wave `wv` owns the
// 16-column strip [wv*16, wv*16+16); produces 8 16x16 f32 accum tiles.
// A layout (16-bit A 16x32): lanes 0-15 M=lane K=0..7/16..23 in v0-3/v4-7;
// lanes 16-31 M=lane-16, K=8..15/24..31.  B layout: lane = N + 16*Khalf,
// v0-3 hold K pairs 8*Khalf+2v, v4-7 hold 16+8*Khalf+2(v-4) (same order as A).
// ---------------------------------------------------------------------------
template<int CHUNKS, int LDA>
static __device__ __forceinline__ void wmma_block_gemm(
    const unsigned int* __restrict__ W,
    const unsigned int* sAp,
    v8f acc[8], int lane, int wv)
{
    const int nloc = lane & 15;
    const int kh   = lane >> 4;
    const int n    = wv * 16 + nloc;
#pragma unroll
    for (int kc = 0; kc < CHUNKS; ++kc) {
        FragAB b;
#pragma unroll
        for (int v = 0; v < 8; ++v)
            b.u[v] = W[(kc * 16 + kh * 4 + (v & 3) + (v >> 2) * 8) * 128 + n];
#pragma unroll
        for (int mt = 0; mt < 8; ++mt) {
            FragAB a;
            const unsigned int* rp = sAp + (mt * 16 + nloc) * LDA + kc * 16 + kh * 4;
#pragma unroll
            for (int v = 0; v < 4; ++v) a.u[v]     = rp[v];
#pragma unroll
            for (int v = 0; v < 4; ++v) a.u[4 + v] = rp[8 + v];
            acc[mt] = __builtin_amdgcn_wmma_f32_16x16x32_bf16(
                false, a.v, false, b.v, (short)0, acc[mt], false, false);
        }
    }
}

// bias + SiLU, store bf16 into LDS [128 x 128] (stride 128 u16)
static __device__ __forceinline__ void store_silu_bf16(
    unsigned short* hS, const v8f acc[8], const float* __restrict__ bias,
    int lane, int wv)
{
    const int nloc = lane & 15, kh = lane >> 4;
    const int n = wv * 16 + nloc;
    const float bv = bias[n];
#pragma unroll
    for (int mt = 0; mt < 8; ++mt) {
#pragma unroll
        for (int i = 0; i < 8; ++i) {
            const int M = mt * 16 + kh * 8 + i;   // D layout: VGPR i -> M = i + 8*kh
            hS[M * 128 + n] = f2bf(siluf(acc[mt][i] + bv));
        }
    }
}

static __device__ __forceinline__ void ln_stats(
    const unsigned short* hS, float* sMu, float* sRs, int tid)
{
    if (tid < 128) {
        float s = 0.f, s2 = 0.f;
#pragma unroll 8
        for (int c = 0; c < 128; ++c) {
            float v = bf2f(hS[tid * 128 + c]);
            s += v; s2 += v * v;
        }
        float mu  = s  * (1.0f / 128.0f);
        float var = s2 * (1.0f / 128.0f) - mu * mu;
        sMu[tid] = mu;
        sRs[tid] = rsqrtf(var + 1e-5f);
    }
}

// in-place LN affine; optionally scatter fp32 result into agg[dst] atomically
static __device__ __forceinline__ void ln_apply(
    unsigned short* hS, const float* sMu, const float* sRs,
    const float* __restrict__ g, const float* __restrict__ b,
    int tid, float* aggp, const int* sDst)
{
#pragma unroll 4
    for (int j = 0; j < 64; ++j) {
        const int idx = tid * 64 + j;          // 256*64 = 16384 elements
        const int r = idx >> 7, c = idx & 127;
        float v = (bf2f(hS[idx]) - sMu[r]) * sRs[r] * g[c] + b[c];
        if (aggp) atomicAdd(&aggp[(size_t)sDst[r] * 128 + c], v);
        hS[idx] = f2bf(v);
    }
}

// ---------------------------------------------------------------------------
// Weight prep: fp32 -> bf16 pair-packed [Kpairs x 128] dwords
// ---------------------------------------------------------------------------
__global__ void prep_weights(
    const float* __restrict__ e_w1, const float* __restrict__ e_w2,
    const float* __restrict__ c_w1, const float* __restrict__ n_w1,
    const float* __restrict__ n_w2,
    unsigned int* __restrict__ W1e, unsigned int* __restrict__ W2e,
    unsigned int* __restrict__ CW1, unsigned int* __restrict__ NW1,
    unsigned int* __restrict__ NW2)
{
    const int idx = blockIdx.x * blockDim.x + threadIdx.x;
    const int S1 = 144 * 128, S2 = S1 + 64 * 128, S3 = S2 + 64 * 128;
    const int S4 = S3 + 128 * 128, S5 = S4 + 64 * 128;
    if (idx < S1) {
        int pr = idx >> 7, n = idx & 127;
        int k0 = 2 * pr, k1 = k0 + 1;
        float a = (k0 < 259) ? e_w1[k0 * 128 + n] : 0.f;
        float b = (k1 < 259) ? e_w1[k1 * 128 + n] : 0.f;
        W1e[idx] = pack2(a, b);
    } else if (idx < S2) {
        int i = idx - S1; int pr = i >> 7, n = i & 127;
        W2e[i] = pack2(e_w2[(2 * pr) * 128 + n], e_w2[(2 * pr + 1) * 128 + n]);
    } else if (idx < S3) {
        int i = idx - S2; int pr = i >> 7, n = i & 127;
        CW1[i] = pack2(c_w1[(2 * pr) * 128 + n], c_w1[(2 * pr + 1) * 128 + n]);
    } else if (idx < S4) {
        int i = idx - S3; int pr = i >> 7, n = i & 127;
        NW1[i] = pack2(n_w1[(2 * pr) * 128 + n], n_w1[(2 * pr + 1) * 128 + n]);
    } else if (idx < S5) {
        int i = idx - S4; int pr = i >> 7, n = i & 127;
        NW2[i] = pack2(n_w2[(2 * pr) * 128 + n], n_w2[(2 * pr + 1) * 128 + n]);
    }
}

__global__ void zero_ws(float* __restrict__ p, size_t n)
{
    size_t i = (size_t)blockIdx.x * blockDim.x + threadIdx.x;
    size_t stride = (size_t)gridDim.x * blockDim.x;
    for (; i < n; i += stride) p[i] = 0.f;
}

// ---------------------------------------------------------------------------
// Edge pipeline: gather -> GEMM1 -> SiLU -> LN -> GEMM2 -> SiLU -> LN (m_ij,
// scatter-add to agg) -> GEMM3 (coord) -> SiLU -> dot(c_w2) -> rel*coef
// scatter-add to posAcc.  One block = 128 edges, 8 waves (wave32).
// ---------------------------------------------------------------------------
__global__ __launch_bounds__(256) void egnn_edge_kernel(
    const float* __restrict__ x, const float* __restrict__ pos,
    const long long* __restrict__ ei, const float* __restrict__ ea,
    const float* __restrict__ e_b1, const float* __restrict__ e_g1,
    const float* __restrict__ e_be1,
    const float* __restrict__ e_b2, const float* __restrict__ e_g2,
    const float* __restrict__ e_be2,
    const float* __restrict__ c_b1, const float* __restrict__ c_w2,
    const float* __restrict__ c_b2,
    const unsigned int* __restrict__ W1e, const unsigned int* __restrict__ W2e,
    const unsigned int* __restrict__ CW1,
    float* __restrict__ agg, float* __restrict__ posAcc)
{
    __shared__ unsigned short sA[128 * 288];   // e_in tile, bf16 (K padded to 288)
    __shared__ unsigned short hS[128 * 128];   // activation staging, bf16
    __shared__ int   sDst[128];
    __shared__ float sRel[128 * 3];
    __shared__ float sMu[128];
    __shared__ float sRs[128];

    const int tid = threadIdx.x, lane = tid & 31, wv = tid >> 5;
    const int ebase = blockIdx.x * 128;
    unsigned int* sAp = (unsigned int*)sA;     // pair view, row stride 144 dwords

    // ---- gather: cols 0-127 x[src], 128-255 x[dst], 256-257 ea, 258 r2, pad 0
    for (int rr = 0; rr < 16; ++rr) {
        const int r = wv * 16 + rr;
        const int e = ebase + r;
        const int src = (int)ei[e];
        const int dst = (int)ei[(size_t)N_EDGES + e];
        const float2* xs = (const float2*)(x + (size_t)src * 128);
        const float2* xd = (const float2*)(x + (size_t)dst * 128);
#pragma unroll
        for (int j = 0; j < 2; ++j) {
            const int p = lane + j * 32;
            float2 v = xs[p];
            sAp[r * 144 + p] = pack2(v.x, v.y);
            float2 w = xd[p];
            sAp[r * 144 + 64 + p] = pack2(w.x, w.y);
        }
        if (lane == 0) {
            sDst[r] = dst;
            float rx = pos[(size_t)dst * 3 + 0] - pos[(size_t)src * 3 + 0];
            float ry = pos[(size_t)dst * 3 + 1] - pos[(size_t)src * 3 + 1];
            float rz = pos[(size_t)dst * 3 + 2] - pos[(size_t)src * 3 + 2];
            sRel[r * 3 + 0] = rx; sRel[r * 3 + 1] = ry; sRel[r * 3 + 2] = rz;
            float r2 = rx * rx + ry * ry + rz * rz;
            sAp[r * 144 + 128] = pack2(ea[(size_t)e * 2 + 0], ea[(size_t)e * 2 + 1]);
            sAp[r * 144 + 129] = pack2(r2, 0.f);
        }
        if (lane >= 2 && lane < 16) sAp[r * 144 + 128 + lane] = 0u;  // K padding
    }
    __syncthreads();

    const v8f vzero = {0.f, 0.f, 0.f, 0.f, 0.f, 0.f, 0.f, 0.f};
    v8f acc[8];

    // ---- GEMM1: [128 x 288] x [288 x 128]
#pragma unroll
    for (int mt = 0; mt < 8; ++mt) acc[mt] = vzero;
    wmma_block_gemm<9, 144>(W1e, sAp, acc, lane, wv);
    store_silu_bf16(hS, acc, e_b1, lane, wv);
    __syncthreads();
    ln_stats(hS, sMu, sRs, tid);
    __syncthreads();
    ln_apply(hS, sMu, sRs, e_g1, e_be1, tid, nullptr, nullptr);
    __syncthreads();

    // ---- GEMM2: [128 x 128] x [128 x 128] -> m_ij (+ scatter to agg)
#pragma unroll
    for (int mt = 0; mt < 8; ++mt) acc[mt] = vzero;
    wmma_block_gemm<4, 64>(W2e, (const unsigned int*)hS, acc, lane, wv);
    __syncthreads();                       // all waves done reading hS
    store_silu_bf16(hS, acc, e_b2, lane, wv);
    __syncthreads();
    ln_stats(hS, sMu, sRs, tid);
    __syncthreads();
    ln_apply(hS, sMu, sRs, e_g2, e_be2, tid, agg, sDst);
    __syncthreads();

    // ---- GEMM3 (coord MLP hidden): ch = silu(m_ij @ c_w1 + c_b1)
#pragma unroll
    for (int mt = 0; mt < 8; ++mt) acc[mt] = vzero;
    wmma_block_gemm<4, 64>(CW1, (const unsigned int*)hS, acc, lane, wv);
    __syncthreads();
    store_silu_bf16(hS, acc, c_b1, lane, wv);
    __syncthreads();

    // ---- coef = ch @ c_w2 + c_b2 ; trans = rel * coef ; scatter to posAcc
    if (tid < 128) {
        const int r = tid;
        float s = 0.f;
#pragma unroll 8
        for (int c = 0; c < 128; ++c) s += bf2f(hS[r * 128 + c]) * c_w2[c];
        const float coef = s + c_b2[0];
        const int d = sDst[r];
        atomicAdd(&posAcc[(size_t)d * 3 + 0], sRel[r * 3 + 0] * coef);
        atomicAdd(&posAcc[(size_t)d * 3 + 1], sRel[r * 3 + 1] * coef);
        atomicAdd(&posAcc[(size_t)d * 3 + 2], sRel[r * 3 + 2] * coef);
    }
}

// ---------------------------------------------------------------------------
// Node pipeline: [x | agg_m] -> GEMM -> SiLU -> LN -> GEMM -> +residual
// Also writes pos_out = pos + posAcc.  One block = 128 nodes.
// ---------------------------------------------------------------------------
__global__ __launch_bounds__(256) void egnn_node_kernel(
    const float* __restrict__ x, const float* __restrict__ pos,
    const float* __restrict__ n_b1, const float* __restrict__ n_g1,
    const float* __restrict__ n_be1, const float* __restrict__ n_b2,
    const unsigned int* __restrict__ NW1, const unsigned int* __restrict__ NW2,
    const float* __restrict__ agg, const float* __restrict__ posAcc,
    float* __restrict__ out)
{
    __shared__ unsigned short sA[128 * 256];
    __shared__ unsigned short hS[128 * 128];
    __shared__ float sMu[128];
    __shared__ float sRs[128];

    const int tid = threadIdx.x, lane = tid & 31, wv = tid >> 5;
    const int nbase = blockIdx.x * 128;
    unsigned int* sAp = (unsigned int*)sA;    // row stride 128 dwords

    for (int rr = 0; rr < 16; ++rr) {
        const int r = wv * 16 + rr;
        const int node = nbase + r;
#pragma unroll
        for (int j = 0; j < 2; ++j) {
            const int p = lane + j * 32;
            if (node < N_NODES) {
                float2 v = ((const float2*)(x + (size_t)node * 128))[p];
                sAp[r * 128 + p] = pack2(v.x, v.y);
                float2 w = ((const float2*)(agg + (size_t)node * 128))[p];
                sAp[r * 128 + 64 + p] = pack2(w.x, w.y);
            } else {
                sAp[r * 128 + p] = 0u;
                sAp[r * 128 + 64 + p] = 0u;
            }
        }
    }
    __syncthreads();

    const v8f vzero = {0.f, 0.f, 0.f, 0.f, 0.f, 0.f, 0.f, 0.f};
    v8f acc[8];

#pragma unroll
    for (int mt = 0; mt < 8; ++mt) acc[mt] = vzero;
    wmma_block_gemm<8, 128>(NW1, sAp, acc, lane, wv);
    store_silu_bf16(hS, acc, n_b1, lane, wv);
    __syncthreads();
    ln_stats(hS, sMu, sRs, tid);
    __syncthreads();
    ln_apply(hS, sMu, sRs, n_g1, n_be1, tid, nullptr, nullptr);
    __syncthreads();

#pragma unroll
    for (int mt = 0; mt < 8; ++mt) acc[mt] = vzero;
    wmma_block_gemm<4, 64>(NW2, (const unsigned int*)hS, acc, lane, wv);

    // residual + bias -> x_out
    {
        const int nloc = lane & 15, kh = lane >> 4;
        const int n = wv * 16 + nloc;
        const float bv = n_b2[n];
#pragma unroll
        for (int mt = 0; mt < 8; ++mt) {
#pragma unroll
            for (int i = 0; i < 8; ++i) {
                const int M = mt * 16 + kh * 8 + i;
                const int node = nbase + M;
                if (node < N_NODES)
                    out[(size_t)node * 128 + n] =
                        x[(size_t)node * 128 + n] + acc[mt][i] + bv;
            }
        }
    }

    // pos_out = pos + posAcc
    if (tid < 128) {
        const int node = nbase + tid;
        if (node < N_NODES) {
#pragma unroll
            for (int k = 0; k < 3; ++k)
                out[OUT_POS_OFF + (size_t)node * 3 + k] =
                    pos[(size_t)node * 3 + k] + posAcc[(size_t)node * 3 + k];
        }
    }
}

// ---------------------------------------------------------------------------
extern "C" void kernel_launch(void* const* d_in, const int* in_sizes, int n_in,
                              void* d_out, int out_size, void* d_ws, size_t ws_size,
                              hipStream_t stream) {
    const float*     x     = (const float*)d_in[0];
    const float*     pos   = (const float*)d_in[1];
    const long long* ei    = (const long long*)d_in[2];
    const float*     ea    = (const float*)d_in[3];
    const float*     e_w1  = (const float*)d_in[4];
    const float*     e_b1  = (const float*)d_in[5];
    const float*     e_g1  = (const float*)d_in[6];
    const float*     e_be1 = (const float*)d_in[7];
    const float*     e_w2  = (const float*)d_in[8];
    const float*     e_b2  = (const float*)d_in[9];
    const float*     e_g2  = (const float*)d_in[10];
    const float*     e_be2 = (const float*)d_in[11];
    const float*     n_w1  = (const float*)d_in[12];
    const float*     n_b1  = (const float*)d_in[13];
    const float*     n_g1  = (const float*)d_in[14];
    const float*     n_be1 = (const float*)d_in[15];
    const float*     n_w2  = (const float*)d_in[16];
    const float*     n_b2  = (const float*)d_in[17];
    const float*     c_w1  = (const float*)d_in[18];
    const float*     c_b1  = (const float*)d_in[19];
    const float*     c_w2  = (const float*)d_in[20];
    const float*     c_b2  = (const float*)d_in[21];

    float* agg    = (float*)d_ws;                          // [N,128] f32
    float* posAcc = agg + (size_t)N_NODES * 128;           // [N,3]   f32
    unsigned int* W1e = (unsigned int*)(posAcc + (size_t)N_NODES * 3);
    unsigned int* W2e = W1e + 144 * 128;
    unsigned int* CW1 = W2e + 64 * 128;
    unsigned int* NW1 = CW1 + 64 * 128;
    unsigned int* NW2 = NW1 + 128 * 128;
    float* out = (float*)d_out;

    prep_weights<<<232, 256, 0, stream>>>(e_w1, e_w2, c_w1, n_w1, n_w2,
                                          W1e, W2e, CW1, NW1, NW2);
    const size_t nz = (size_t)N_NODES * 128 + (size_t)N_NODES * 3;
    zero_ws<<<1024, 256, 0, stream>>>(agg, nz);

    egnn_edge_kernel<<<N_EDGES / 128, 256, 0, stream>>>(
        x, pos, ei, ea, e_b1, e_g1, e_be1, e_b2, e_g2, e_be2,
        c_b1, c_w2, c_b2, W1e, W2e, CW1, agg, posAcc);

    egnn_node_kernel<<<(N_NODES + 127) / 128, 256, 0, stream>>>(
        x, pos, n_b1, n_g1, n_be1, n_b2, NW1, NW2, agg, posAcc, out);
}
